// GINVirtual_node_46909632806968
// MI455X (gfx1250) — matrix operand order repaired and checked by reading, not compiled, not run on hardware
//
#include <hip/hip_runtime.h>
#include <hip/hip_bf16.h>

#define NN 50000
#define EE 600000
#define DD 128
#define LL 3
#define GG 500
#define AVV 128
#define BVV 8

typedef _Float16 h16 __attribute__((ext_vector_type(16)));
typedef _Float16 h8v __attribute__((ext_vector_type(8)));
typedef float    f8  __attribute__((ext_vector_type(8)));
typedef float    f4v __attribute__((ext_vector_type(4)));

// ---------------------------------------------------------------- utilities
__global__ void k_zero(float* p, int n) {
    int i = blockIdx.x * blockDim.x + threadIdx.x;
    if (i < n) p[i] = 0.f;
}

__global__ void k_copy(float* dst, const float* src, int n) {
    int i = blockIdx.x * blockDim.x + threadIdx.x;
    if (i < n) dst[i] = src[i];
}

// o[m,k] = (f16) w[k,m]   (transpose so B fragments load contiguously)
__global__ void k_f2h_T(const float* __restrict__ w, _Float16* __restrict__ o, int K,
                        int M) {
    int i = blockIdx.x * blockDim.x + threadIdx.x;
    if (i >= K * M) return;
    int k = i / M, m = i % M;
    o[(size_t)m * K + k] = (_Float16)w[i];
}

// vn[g,:] = vn_emb[0,:]
__global__ void k_bcast(const float* __restrict__ vemb, float* __restrict__ vn) {
    vn[blockIdx.x * DD + threadIdx.x] = vemb[threadIdx.x];
}

// ---------------------------------------------------------------- encoders
__global__ void k_atom_enc(const int* __restrict__ x, const float* __restrict__ aemb,
                           float* __restrict__ h) {
    int n = blockIdx.x, d = threadIdx.x;
    float acc = 0.f;
#pragma unroll
    for (int i = 0; i < 9; ++i) {
        int v = x[n * 9 + i];
        acc += aemb[((size_t)i * AVV + v) * DD + d];
    }
    h[(size_t)n * DD + d] = acc;
}

__global__ void k_add_vn(float* __restrict__ h, const float* __restrict__ vn,
                         const int* __restrict__ batch) {
    int n = blockIdx.x, d = threadIdx.x;
    h[(size_t)n * DD + d] += vn[(size_t)batch[n] * DD + d];
}

// msg = relu(h[src] + bond_emb), scattered into agg[dst]. 32 lanes x float4 per edge.
__global__ void k_edge_scatter(const float* __restrict__ h, const int* __restrict__ ei,
                               const int* __restrict__ ea, const float* __restrict__ bemb,
                               float* __restrict__ agg) {
    int e = blockIdx.x * 8 + (threadIdx.x >> 5);
    int q = (threadIdx.x & 31) * 4;
    if (e >= EE) return;
    int s = ei[e];
    int t = ei[EE + e];
    if (q == 0) __builtin_prefetch(&agg[(size_t)t * DD], 0, 1);  // global_prefetch_b8
    f4v ee = {};
#pragma unroll
    for (int i = 0; i < 3; ++i)
        ee += *(const f4v*)&bemb[((size_t)i * BVV + ea[e * 3 + i]) * DD + q];
    f4v m = *(const f4v*)&h[(size_t)s * DD + q] + ee;
    float* ap = &agg[(size_t)t * DD + q];
#pragma unroll
    for (int j = 0; j < 4; ++j) {
        float v = m[j] > 0.f ? m[j] : 0.f;
        atomicAdd(ap + j, v);
    }
}

// pooled: vt[batch[n],:] += h[n,:]
__global__ void k_pool(const float* __restrict__ h, const int* __restrict__ batch,
                       float* __restrict__ vt) {
    int n = blockIdx.x, d = threadIdx.x;
    atomicAdd(&vt[(size_t)batch[n] * DD + d], h[(size_t)n * DD + d]);
}

// ---------------------------------------------------------------- WMMA GEMMs
// One A fragment per K-step, NT=4 column tiles per wave, 4 waves -> full M=256.
// A = (1+eps)*h + agg  [N x 128] f32 -> f16 on the fly; Wt = W1^T [256 x 128] f16.
__global__ void k_gemm1(const float* __restrict__ h, const float* __restrict__ agg,
                        const float* __restrict__ eps, int l,
                        const _Float16* __restrict__ Wt, const float* __restrict__ bias,
                        float* __restrict__ out, float* __restrict__ csum,
                        float* __restrict__ csq) {
    const int lane = threadIdx.x & 31;
    const int wave = threadIdx.x >> 5;
    const int hs = lane >> 4;
    const int ln = lane & 15;
    const int K = DD, M = 2 * DD;
    const int row0 = blockIdx.x * 16;
    const int m = row0 + ln;
    const float e1 = 1.f + eps[l];
    f8 acc[4] = {};
#pragma unroll
    for (int kb = 0; kb < K; kb += 32) {
        // A fragment: two contiguous 8-float groups at kb+8*hs and kb+16+8*hs
        size_t b0 = (size_t)m * K + kb + 8 * hs;
        f4v h0 = *(const f4v*)(h + b0), h1 = *(const f4v*)(h + b0 + 4);
        f4v g0 = *(const f4v*)(agg + b0), g1 = *(const f4v*)(agg + b0 + 4);
        f4v h2 = *(const f4v*)(h + b0 + 16), h3 = *(const f4v*)(h + b0 + 20);
        f4v g2 = *(const f4v*)(agg + b0 + 16), g3 = *(const f4v*)(agg + b0 + 20);
        h16 a;
#pragma unroll
        for (int j = 0; j < 4; ++j) {
            a[j]      = (_Float16)(e1 * h0[j] + g0[j]);
            a[4 + j]  = (_Float16)(e1 * h1[j] + g1[j]);
            a[8 + j]  = (_Float16)(e1 * h2[j] + g2[j]);
            a[12 + j] = (_Float16)(e1 * h3[j] + g3[j]);
        }
#pragma unroll
        for (int t = 0; t < 4; ++t) {
            int c = wave * 16 + t * 64 + ln;
            const h8v* wp = (const h8v*)&Wt[(size_t)c * K + kb + 8 * hs];
            h8v w0 = wp[0], w1 = wp[2];
            h16 b;
#pragma unroll
            for (int j = 0; j < 8; ++j) {
                b[j] = w0[j];
                b[8 + j] = w1[j];
            }
            acc[t] = __builtin_amdgcn_wmma_f32_16x16x32_f16(false, a, false, b, (short)0,
                                                            acc[t], false, false);
        }
    }
#pragma unroll
    for (int t = 0; t < 4; ++t) {
        int c = wave * 16 + t * 64 + ln;
        float bi = bias[c];
        float s = 0.f, sq = 0.f;
#pragma unroll
        for (int r = 0; r < 8; ++r) {
            float v = acc[t][r] + bi;
            out[(size_t)(row0 + r + 8 * hs) * M + c] = v;
            s += v;
            sq += v * v;
        }
        atomicAdd(&csum[c], s);
        atomicAdd(&csq[c], sq);
    }
}

// A = relu(BN(t1)) via fused scale/shift [N x 256]; Wt = W2^T [128 x 256] f16. NT=2.
__global__ void k_gemm2(const float* __restrict__ t1, const float* __restrict__ sc,
                        const float* __restrict__ sh, const _Float16* __restrict__ Wt,
                        const float* __restrict__ bias, float* __restrict__ out,
                        float* __restrict__ csum, float* __restrict__ csq) {
    const int lane = threadIdx.x & 31;
    const int wave = threadIdx.x >> 5;
    const int hs = lane >> 4;
    const int ln = lane & 15;
    const int K = 2 * DD, M = DD;
    const int row0 = blockIdx.x * 16;
    const int m = row0 + ln;
    f8 acc[2] = {};
#pragma unroll
    for (int kb = 0; kb < K; kb += 32) {
        int k0 = kb + 8 * hs;
        size_t b0 = (size_t)m * K + k0;
        f4v t0 = *(const f4v*)(t1 + b0), t1v = *(const f4v*)(t1 + b0 + 4);
        f4v t2 = *(const f4v*)(t1 + b0 + 16), t3 = *(const f4v*)(t1 + b0 + 20);
        f4v s0 = *(const f4v*)(sc + k0), s1 = *(const f4v*)(sc + k0 + 4);
        f4v s2 = *(const f4v*)(sc + k0 + 16), s3 = *(const f4v*)(sc + k0 + 20);
        f4v d0 = *(const f4v*)(sh + k0), d1 = *(const f4v*)(sh + k0 + 4);
        f4v d2 = *(const f4v*)(sh + k0 + 16), d3 = *(const f4v*)(sh + k0 + 20);
        h16 a;
#pragma unroll
        for (int j = 0; j < 4; ++j) {
            float v0 = t0[j] * s0[j] + d0[j];
            float v1 = t1v[j] * s1[j] + d1[j];
            float v2 = t2[j] * s2[j] + d2[j];
            float v3 = t3[j] * s3[j] + d3[j];
            a[j]      = (_Float16)(v0 > 0.f ? v0 : 0.f);
            a[4 + j]  = (_Float16)(v1 > 0.f ? v1 : 0.f);
            a[8 + j]  = (_Float16)(v2 > 0.f ? v2 : 0.f);
            a[12 + j] = (_Float16)(v3 > 0.f ? v3 : 0.f);
        }
#pragma unroll
        for (int t = 0; t < 2; ++t) {
            int c = wave * 16 + t * 64 + ln;
            const h8v* wp = (const h8v*)&Wt[(size_t)c * K + kb + 8 * hs];
            h8v w0 = wp[0], w1 = wp[2];
            h16 b;
#pragma unroll
            for (int j = 0; j < 8; ++j) {
                b[j] = w0[j];
                b[8 + j] = w1[j];
            }
            acc[t] = __builtin_amdgcn_wmma_f32_16x16x32_f16(false, a, false, b, (short)0,
                                                            acc[t], false, false);
        }
    }
#pragma unroll
    for (int t = 0; t < 2; ++t) {
        int c = wave * 16 + t * 64 + ln;
        float bi = bias[c];
        float s = 0.f, sq = 0.f;
#pragma unroll
        for (int r = 0; r < 8; ++r) {
            float v = acc[t][r] + bi;
            out[(size_t)(row0 + r + 8 * hs) * M + c] = v;
            s += v;
            sq += v * v;
        }
        atomicAdd(&csum[c], s);
        atomicAdd(&csq[c], sq);
    }
}

// ---------------------------------------------------------------- BN helpers
__global__ void k_finalize(const float* __restrict__ sum, const float* __restrict__ sq,
                           const float* __restrict__ g, const float* __restrict__ b,
                           float* __restrict__ sc, float* __restrict__ sh, int M,
                           float invc) {
    int c = blockIdx.x * blockDim.x + threadIdx.x;
    if (c >= M) return;
    float mu = sum[c] * invc;
    float var = sq[c] * invc - mu * mu;
    float rs = rsqrtf(var + 1e-5f);
    float scale = g[c] * rs;
    sc[c] = scale;
    sh[c] = b[c] - mu * scale;
}

__global__ void k_apply(const float* __restrict__ t, const float* __restrict__ sc,
                        const float* __restrict__ sh, float* __restrict__ o, int M,
                        int relu) {
    int n = blockIdx.x, d = threadIdx.x;
    float v = t[(size_t)n * M + d] * sc[d] + sh[d];
    if (relu) v = v > 0.f ? v : 0.f;
    o[(size_t)n * M + d] = v;
}

// ---------------------------------------------------------------- small GEMM (virtual node)
__global__ void k_sgemm(const float* __restrict__ A, const float* __restrict__ sc,
                        const float* __restrict__ sh, int affine,
                        const float* __restrict__ W, const float* __restrict__ bias,
                        float* __restrict__ out, float* __restrict__ csum,
                        float* __restrict__ csq, int K, int M) {
    __shared__ float As[256];
    int r = blockIdx.x, m = threadIdx.x;
    for (int k = m; k < K; k += blockDim.x) {
        float v = A[(size_t)r * K + k];
        if (affine) {
            v = v * sc[k] + sh[k];
            v = v > 0.f ? v : 0.f;
        }
        As[k] = v;
    }
    __syncthreads();
    float acc = bias[m];
    for (int k = 0; k < K; ++k) acc += As[k] * W[(size_t)k * M + m];
    out[(size_t)r * M + m] = acc;
    atomicAdd(&csum[m], acc);
    atomicAdd(&csq[m], acc * acc);
}

// ---------------------------------------------------------------- host
extern "C" void kernel_launch(void* const* d_in, const int* in_sizes, int n_in,
                              void* d_out, int out_size, void* d_ws, size_t ws_size,
                              hipStream_t stream) {
    const int*   x     = (const int*)d_in[0];
    const int*   ei    = (const int*)d_in[1];
    const int*   ea    = (const int*)d_in[2];
    const int*   batch = (const int*)d_in[3];
    const float* aemb  = (const float*)d_in[4];
    const float* bemb  = (const float*)d_in[5];
    const float* eps   = (const float*)d_in[6];
    const float* cw1   = (const float*)d_in[7];
    const float* cb1   = (const float*)d_in[8];
    const float* cbng  = (const float*)d_in[9];
    const float* cbnb  = (const float*)d_in[10];
    const float* cw2   = (const float*)d_in[11];
    const float* cb2   = (const float*)d_in[12];
    const float* bng   = (const float*)d_in[13];
    const float* bnb   = (const float*)d_in[14];
    const float* vemb  = (const float*)d_in[15];
    const float* vnw1  = (const float*)d_in[16];
    const float* vnb1  = (const float*)d_in[17];
    const float* v1g   = (const float*)d_in[18];
    const float* v1b   = (const float*)d_in[19];
    const float* vnw2  = (const float*)d_in[20];
    const float* vnb2  = (const float*)d_in[21];
    const float* v2g   = (const float*)d_in[22];
    const float* v2b   = (const float*)d_in[23];
    (void)in_sizes; (void)n_in; (void)out_size; (void)ws_size;

    float* ws = (float*)d_ws;
    size_t off = 0;
    float* hA  = ws + off; off += (size_t)NN * DD;
    float* hB  = ws + off; off += (size_t)NN * DD;
    float* agg = ws + off; off += (size_t)NN * DD;       // reused as GEMM2 out
    float* t1  = ws + off; off += (size_t)NN * 2 * DD;
    float* vn  = ws + off; off += (size_t)GG * DD;
    float* vt  = ws + off; off += (size_t)GG * DD;
    float* vt1 = ws + off; off += (size_t)GG * 2 * DD;
    float* vt2 = ws + off; off += (size_t)GG * DD;
    float* st  = ws + off; off += (size_t)16 * 256;
    float* sumA = st, *sqA = st + 256, *scA = st + 512, *shA = st + 768;
    float* sumB = st + 1024, *sqB = st + 1280, *scB = st + 1536, *shB = st + 1792;
    float* sumC = st + 2048, *sqC = st + 2304, *scC = st + 2560, *shC = st + 2816;
    float* sumD = st + 3072, *sqD = st + 3328, *scD = st + 3584, *shD = st + 3840;
    _Float16* W1h = (_Float16*)(ws + off); off += (size_t)DD * 2 * DD / 2;  // W1^T
    _Float16* W2h = (_Float16*)(ws + off); off += (size_t)DD * 2 * DD / 2;  // W2^T

    const int ND = NN * DD;
    auto g1 = [](int n) { return (n + 255) / 256; };

    k_atom_enc<<<NN, DD, 0, stream>>>(x, aemb, hA);
    k_bcast<<<GG, DD, 0, stream>>>(vemb, vn);

    float* hcur = hA;
    float* hnext = hB;
    for (int l = 0; l < LL; ++l) {
        // message passing
        k_zero<<<g1(ND), 256, 0, stream>>>(agg, ND);
        k_add_vn<<<NN, DD, 0, stream>>>(hcur, vn, batch);
        k_edge_scatter<<<EE / 8, 256, 0, stream>>>(hcur, ei, ea,
                                                   bemb + (size_t)l * 3 * BVV * DD, agg);
        // weights -> f16, transposed
        k_f2h_T<<<g1(DD * 2 * DD), 256, 0, stream>>>(cw1 + (size_t)l * DD * 2 * DD, W1h,
                                                     DD, 2 * DD);
        k_f2h_T<<<g1(2 * DD * DD), 256, 0, stream>>>(cw2 + (size_t)l * 2 * DD * DD, W2h,
                                                     2 * DD, DD);
        // MLP stage 1 (WMMA) + fused BN stats
        k_zero<<<2, 256, 0, stream>>>(sumA, 512);
        k_gemm1<<<NN / 16, 128, 0, stream>>>(hcur, agg, eps, l, W1h,
                                             cb1 + (size_t)l * 2 * DD, t1, sumA, sqA);
        k_finalize<<<1, 2 * DD, 0, stream>>>(sumA, sqA, cbng + (size_t)l * 2 * DD,
                                             cbnb + (size_t)l * 2 * DD, scA, shA, 2 * DD,
                                             1.f / NN);
        // MLP stage 2 (WMMA, BN+relu fused in prolog) + fused BN stats
        k_zero<<<2, 256, 0, stream>>>(sumB, 512);
        k_gemm2<<<NN / 16, 128, 0, stream>>>(t1, scA, shA, W2h, cb2 + (size_t)l * DD, agg,
                                             sumB, sqB);
        k_finalize<<<1, DD, 0, stream>>>(sumB, sqB, bng + (size_t)l * DD,
                                         bnb + (size_t)l * DD, scB, shB, DD, 1.f / NN);

        if (l < LL - 1) {
            k_apply<<<NN, DD, 0, stream>>>(agg, scB, shB, hnext, DD, 1);
            // virtual node update
            k_copy<<<g1(GG * DD), 256, 0, stream>>>(vt, vn, GG * DD);
            k_pool<<<NN, DD, 0, stream>>>(hcur, batch, vt);
            k_zero<<<2, 256, 0, stream>>>(sumC, 512);
            k_sgemm<<<GG, 2 * DD, 0, stream>>>(vt, scC, shC, 0,
                                               vnw1 + (size_t)l * DD * 2 * DD,
                                               vnb1 + (size_t)l * 2 * DD, vt1, sumC, sqC,
                                               DD, 2 * DD);
            k_finalize<<<1, 2 * DD, 0, stream>>>(sumC, sqC, v1g + (size_t)l * 2 * DD,
                                                 v1b + (size_t)l * 2 * DD, scC, shC,
                                                 2 * DD, 1.f / GG);
            k_zero<<<2, 256, 0, stream>>>(sumD, 512);
            k_sgemm<<<GG, DD, 0, stream>>>(vt1, scC, shC, 1,
                                           vnw2 + (size_t)l * 2 * DD * DD,
                                           vnb2 + (size_t)l * DD, vt2, sumD, sqD, 2 * DD,
                                           DD);
            k_finalize<<<1, DD, 0, stream>>>(sumD, sqD, v2g + (size_t)l * DD,
                                             v2b + (size_t)l * DD, scD, shD, DD,
                                             1.f / GG);
            k_apply<<<GG, DD, 0, stream>>>(vt2, scD, shD, vn, DD, 1);
            float* tmp = hcur; hcur = hnext; hnext = tmp;
        } else {
            k_apply<<<NN, DD, 0, stream>>>(agg, scB, shB, (float*)d_out, DD, 0);
        }
    }
}
